// ScaledDotProductAttention_65292092834297
// MI455X (gfx1250) — matrix operand order repaired and checked by reading, not compiled, not run on hardware
//
#include <hip/hip_runtime.h>
#include <hip/hip_bf16.h>

// ---------------------------------------------------------------------------
// Multi-head attention forward for MI455X (gfx1250, wave32, WMMA + TDM).
// B=2, NQ=NK=2048, D_MODEL=1024, H=8, DK=DV=128.
//
// Roofline: ~70 GFLOP of matmul vs ~0.55 GB mandatory HBM traffic
// (attn_weights f32 + attn_mask). Flash-attention fusion keeps the
// 2048x2048 score matrices in registers/LDS; all matmuls run on
// v_wmma_f32_16x16x32_f16; K tiles are staged LDS-side by the Tensor Data
// Mover (tensor_load_to_lds + s_wait_tensorcnt) so the vector pipes stay
// free for WMMA.
// ---------------------------------------------------------------------------

#define BB  2
#define NQS 2048
#define NKS 2048
#define DM  1024
#define HH  8
#define DKh 128
#define DVh 128

typedef __attribute__((ext_vector_type(16))) _Float16 v16h;
typedef __attribute__((ext_vector_type(8)))  _Float16 v8h;
typedef __attribute__((ext_vector_type(4)))  _Float16 v4h;
typedef __attribute__((ext_vector_type(8)))  float    v8f;
typedef __attribute__((ext_vector_type(4)))  float    v4f;
typedef __attribute__((ext_vector_type(4)))  unsigned int v4u;
typedef __attribute__((ext_vector_type(8)))  unsigned int v8u;

__device__ __forceinline__ v8f zero8() {
  v8f z = {0.f, 0.f, 0.f, 0.f, 0.f, 0.f, 0.f, 0.f};
  return z;
}

__device__ __forceinline__ v8f wmma16(v16h a, v16h b, v8f c) {
  // (neg_a, A, neg_b, B, c_mod, C, reuse_a, reuse_b)
  return __builtin_amdgcn_wmma_f32_16x16x32_f16(false, a, false, b, (short)0, c,
                                                false, false);
}

// A-fragment (16x32 f16, M x K). Source layout: row-major [M][K], stride in
// halves. ISA 7.12.2: lanes 0-15 hold row M=lane, K={0..7,16..23};
// lanes 16-31 hold row M=lane-16, K={8..15,24..31}.
__device__ __forceinline__ v16h frag_a(const _Float16* base, int stride, int lane) {
  const _Float16* p = base + (lane & 15) * stride + ((lane & 16) ? 8 : 0);
  v8h lo = *(const v8h*)(p);        // K koff..koff+7
  v8h hi = *(const v8h*)(p + 16);   // K 16+koff..16+koff+7
  return __builtin_shufflevector(lo, hi, 0, 1, 2, 3, 4, 5, 6, 7,
                                 8, 9, 10, 11, 12, 13, 14, 15);
}

// B-fragment (32x16 f16, K x N). Source layout: [N][K] (K contiguous),
// stride in halves. Lanes 0-15: col N=lane, K={0..7,16..23}; lanes 16-31: K={8..15,24..31}.
__device__ __forceinline__ v16h frag_b(const _Float16* base, int stride, int lane) {
  const _Float16* p = base + (lane & 15) * stride + ((lane & 16) ? 8 : 0);
  v8h lo = *(const v8h*)(p);        // K koff..koff+7
  v8h hi = *(const v8h*)(p + 16);   // K 16+koff..16+koff+7
  return __builtin_shufflevector(lo, hi, 0, 1, 2, 3, 4, 5, 6, 7,
                                 8, 9, 10, 11, 12, 13, 14, 15);
}

// ---------------------------------------------------------------------------
// GEMM: C(4096x1024) = X(4096x1024,f32) @ W(1024x1024,f32) + bias.
// MODE 0: store f16 head-major [b][h][tok][d] (for attention).
// MODE 1: store f32 row-major [m][n] (final output projection).
// Block = 256 threads = 8 waves; tile 128x128, K-step 32.
// ---------------------------------------------------------------------------
template <int MODE>
__global__ __launch_bounds__(256) void gemm_kernel(
    const float* __restrict__ X, const float* __restrict__ W,
    const float* __restrict__ bias, _Float16* __restrict__ outh,
    float* __restrict__ outf) {
  __shared__ __attribute__((aligned(16))) _Float16 lA[128 * 48];  // [m][k], pad 48
  __shared__ __attribute__((aligned(16))) _Float16 lB[128 * 48];  // [n][k], pad 48

  const int tid  = threadIdx.x;
  const int lane = tid & 31;
  const int wid  = tid >> 5;
  const int wm   = wid & 3;   // 4 M-quarters of 32 rows
  const int wn   = wid >> 2;  // 2 N-halves of 64 cols
  const int m0   = blockIdx.y * 128;
  const int n0   = blockIdx.x * 128;

  v8f acc[2][4];
#pragma unroll
  for (int i = 0; i < 2; ++i)
#pragma unroll
    for (int j = 0; j < 4; ++j) acc[i][j] = zero8();

  for (int k0 = 0; k0 < DM; k0 += 32) {
    // Stage A tile 128x32 (f32 -> f16)
#pragma unroll
    for (int rr = 0; rr < 4; ++rr) {
      int row = rr * 32 + (tid >> 3);
      int col = (tid & 7) * 4;
      v4f x = *(const v4f*)(X + (size_t)(m0 + row) * DM + k0 + col);
      v4h hx = {(_Float16)x.x, (_Float16)x.y, (_Float16)x.z, (_Float16)x.w};
      *(v4h*)(lA + row * 48 + col) = hx;
    }
    // Stage B tile 32x128 transposed -> lB[n][k] (f32 -> f16)
#pragma unroll
    for (int rr = 0; rr < 4; ++rr) {
      int k = rr * 8 + (tid >> 5);
      int n = (tid & 31) * 4;
      v4f w = *(const v4f*)(W + (size_t)(k0 + k) * DM + n0 + n);
      lB[(n + 0) * 48 + k] = (_Float16)w.x;
      lB[(n + 1) * 48 + k] = (_Float16)w.y;
      lB[(n + 2) * 48 + k] = (_Float16)w.z;
      lB[(n + 3) * 48 + k] = (_Float16)w.w;
    }
    __syncthreads();

    v16h af[2], bf[4];
#pragma unroll
    for (int mi = 0; mi < 2; ++mi)
      af[mi] = frag_a(lA + (wm * 32 + mi * 16) * 48, 48, lane);
#pragma unroll
    for (int ni = 0; ni < 4; ++ni)
      bf[ni] = frag_b(lB + (wn * 64 + ni * 16) * 48, 48, lane);
#pragma unroll
    for (int mi = 0; mi < 2; ++mi)
#pragma unroll
      for (int ni = 0; ni < 4; ++ni)
        acc[mi][ni] = wmma16(af[mi], bf[ni], acc[mi][ni]);
    __syncthreads();
  }

  // C layout (ISA): VGPR r -> M = r + (lane>=16 ? 8 : 0), N = lane&15.
  const int half8 = (lane & 16) ? 8 : 0;
#pragma unroll
  for (int ni = 0; ni < 4; ++ni) {
    int n = n0 + wn * 64 + ni * 16 + (lane & 15);
    float bv = bias[n];
#pragma unroll
    for (int mi = 0; mi < 2; ++mi) {
#pragma unroll
      for (int r = 0; r < 8; ++r) {
        int m = m0 + wm * 32 + mi * 16 + r + half8;
        float v = acc[mi][ni][r] + bv;
        if (MODE == 0) {
          int bb = m >> 11, tok = m & 2047;
          int h = n >> 7, d = n & 127;
          outh[(((size_t)(bb * HH + h)) * NQS + tok) * DKh + d] = (_Float16)v;
        } else {
          outf[(size_t)m * DM + n] = v;
        }
      }
    }
  }
}

// ---------------------------------------------------------------------------
// Flash attention: per block = 128 queries of one (b,h); 8 waves x 16 rows.
// K tile staged by the Tensor Data Mover (one tensor_load_to_lds per tile,
// pad_enable handles the 136-half LDS stride); V tile transposed manually.
// S and P*V via WMMA; attn_weights/mask streamed straight from HBM.
// ---------------------------------------------------------------------------
__global__ __launch_bounds__(256) void attn_kernel(
    const _Float16* __restrict__ Q, const _Float16* __restrict__ K,
    const _Float16* __restrict__ V, const float* __restrict__ AW,
    const int* __restrict__ MSK, float* __restrict__ ctx) {
  __shared__ __attribute__((aligned(16))) _Float16 lK[64 * 136];   // [kv][d]
  __shared__ __attribute__((aligned(16))) _Float16 lVt[128 * 72];  // [dv][kv]
  __shared__ __attribute__((aligned(16))) _Float16 lP[8][16 * 72]; // per wave

  const int tid  = threadIdx.x;
  const int lane = tid & 31;
  const int wid  = tid >> 5;
  const int h = blockIdx.y, b = blockIdx.z;
  const int bh = b * HH + h;
  const int qw = blockIdx.x * 128 + wid * 16;  // this wave's 16 query rows
  const int half8 = (lane & 16) ? 8 : 0;
  const float scale = 0.08838834764831845f;  // 1/sqrt(128)

  // LDS byte offset of lK for the TDM descriptor (addrspace(3) offset is the
  // low 32 bits of the generic shared-aperture address).
  const unsigned lkBase = (unsigned)(uintptr_t)(&lK[0]);

  // Persistent Q A-fragments (16 x 128 = 4 frags of 16x32), straight from HBM.
  const _Float16* qbase = Q + ((size_t)bh * NQS + qw) * DKh;
  v16h qf[4];
#pragma unroll
  for (int f = 0; f < 4; ++f) qf[f] = frag_a(qbase + f * 32, DKh, lane);

  v8f o[8];  // O accumulator: 16 rows x 128 dv
#pragma unroll
  for (int f = 0; f < 8; ++f) o[f] = zero8();
  float mrun[8], lrun[8];
#pragma unroll
  for (int r = 0; r < 8; ++r) { mrun[r] = -1e30f; lrun[r] = 0.f; }

  for (int kv0 = 0; kv0 < NKS; kv0 += 64) {
    // --- K tile 64x128 f16 via Tensor Data Mover (wave 0 issues) ----------
    if (wid == 0) {
      unsigned long long ga =
          (unsigned long long)(uintptr_t)(K + ((size_t)bh * NKS + kv0) * DKh);
      // D# group 0: count=1 (valid), lds_addr, global_addr[56:0], type=2.
      v4u g0 = {1u, lkBase, (unsigned)ga,
                (unsigned)((ga >> 32) & 0x01FFFFFFu) | (2u << 30)};
      // D# group 1:
      //  w0: wg_mask=0 | data_size=1 (2B)<<16 | pad_enable<<20 |
      //      pad_interval=5 (64 DW = one 256B row)<<22 | pad_amount=3 (4 DW)<<25
      //  tensor_dim0=128 (bits 79:48), tensor_dim1=2048 (bits 111:80),
      //  tile_dim0=128 (bits 127:112), tile_dim1=64 (bits 143:128),
      //  tile_dim2=0, tensor_dim0_stride=128 (bits 207:160), dim1_stride=0.
      v8u g1 = {(1u << 16) | (1u << 20) | (5u << 22) | (3u << 25),
                (128u << 16),
                (2048u << 16),
                (128u << 16),
                64u,
                128u,
                0u,
                0u};
      asm volatile("tensor_load_to_lds %0, %1" : : "s"(g0), "s"(g1) : "memory");
    }

    // --- V tile transposed [dv][kv] (manual: TDM cannot transpose) --------
#pragma unroll
    for (int i = 0; i < 4; ++i) {
      int c = tid + i * 256;
      int row = c >> 4, col = (c & 15) * 8;
      v8h vv = *(const v8h*)(V + ((size_t)bh * NKS + kv0 + row) * DVh + col);
#pragma unroll
      for (int j = 0; j < 8; ++j) lVt[(col + j) * 72 + row] = vv[j];
    }
    if (wid == 0) __builtin_amdgcn_s_wait_tensorcnt(0);
    __syncthreads();

    // S (16 x 64) = Q (16x128) . K^T: A=Q frags, B from lK[kv][d].
    v8f s[4];
#pragma unroll
    for (int ni = 0; ni < 4; ++ni) {
      s[ni] = zero8();
#pragma unroll
      for (int f = 0; f < 4; ++f) {
        v16h kb = frag_b(lK + (ni * 16) * 136 + f * 32, 136, lane);
        s[ni] = wmma16(qf[f], kb, s[ni]);
      }
    }

    // scale * attn_weights, then mask -> -inf (the mandatory HBM stream).
#pragma unroll
    for (int ni = 0; ni < 4; ++ni) {
      int kv = kv0 + ni * 16 + (lane & 15);
#pragma unroll
      for (int r = 0; r < 8; ++r) {
        size_t idx = ((size_t)bh * NQS + (qw + r + half8)) * (size_t)NKS + kv;
        float a = s[ni][r] * scale * AW[idx];
        s[ni][r] = (MSK[idx] != 0) ? -1e30f : a;
      }
    }

    // Online softmax; row stats reduce across the 16 lanes of each half-wave.
    float mnew[8], psum[8];
#pragma unroll
    for (int r = 0; r < 8; ++r) {
      float mx = mrun[r];
#pragma unroll
      for (int ni = 0; ni < 4; ++ni) mx = fmaxf(mx, s[ni][r]);
#pragma unroll
      for (int off = 1; off < 16; off <<= 1) mx = fmaxf(mx, __shfl_xor(mx, off, 32));
      mnew[r] = mx;
    }
#pragma unroll
    for (int r = 0; r < 8; ++r) {
      float sum = 0.f;
#pragma unroll
      for (int ni = 0; ni < 4; ++ni) {
        float p = __expf(s[ni][r] - mnew[r]);
        s[ni][r] = p;
        sum += p;
      }
#pragma unroll
      for (int off = 1; off < 16; off <<= 1) sum += __shfl_xor(sum, off, 32);
      psum[r] = sum;
    }
#pragma unroll
    for (int r = 0; r < 8; ++r) {
      float alpha = __expf(mrun[r] - mnew[r]);
      mrun[r] = mnew[r];
      lrun[r] = lrun[r] * alpha + psum[r];
#pragma unroll
      for (int f = 0; f < 8; ++f) o[f][r] *= alpha;
    }

    // C-layout P -> per-wave LDS scratch -> reload as A-fragments (f16).
    _Float16* myP = lP[wid];
#pragma unroll
    for (int ni = 0; ni < 4; ++ni)
#pragma unroll
      for (int r = 0; r < 8; ++r)
        myP[(r + half8) * 72 + ni * 16 + (lane & 15)] = (_Float16)s[ni][r];

    v16h pf0 = frag_a(myP, 72, lane);        // kv 0..31
    v16h pf1 = frag_a(myP + 32, 72, lane);   // kv 32..63

    // O += P (16x64) . V (64x128): B from lVt[dv][kv].
#pragma unroll
    for (int f = 0; f < 8; ++f) {
      v16h vb0 = frag_b(lVt + (f * 16) * 72, 72, lane);
      v16h vb1 = frag_b(lVt + (f * 16) * 72 + 32, 72, lane);
      o[f] = wmma16(pf0, vb0, o[f]);
      o[f] = wmma16(pf1, vb1, o[f]);
    }
    __syncthreads();
  }

  // Normalize and write context f32 [b][q][h*128+dv] for the output GEMM.
  float linv[8];
#pragma unroll
  for (int r = 0; r < 8; ++r) linv[r] = 1.0f / lrun[r];
#pragma unroll
  for (int f = 0; f < 8; ++f) {
    int dv = f * 16 + (lane & 15);
#pragma unroll
    for (int r = 0; r < 8; ++r) {
      int q = qw + r + half8;
      ctx[((size_t)b * NQS + q) * (HH * DVh) + h * DVh + dv] = o[f][r] * linv[r];
    }
  }
}

// ---------------------------------------------------------------------------
extern "C" void kernel_launch(void* const* d_in, const int* in_sizes, int n_in,
                              void* d_out, int out_size, void* d_ws, size_t ws_size,
                              hipStream_t stream) {
  (void)in_sizes; (void)n_in; (void)out_size; (void)ws_size;
  const float* queries = (const float*)d_in[0];
  const float* keys    = (const float*)d_in[1];
  const float* values  = (const float*)d_in[2];
  const int*   amask   = (const int*)d_in[3];
  const float* aweight = (const float*)d_in[4];
  const float* Wq = (const float*)d_in[5];
  const float* bq = (const float*)d_in[6];
  const float* Wk = (const float*)d_in[7];
  const float* bk = (const float*)d_in[8];
  const float* Wv = (const float*)d_in[9];
  const float* bv = (const float*)d_in[10];
  const float* Wo = (const float*)d_in[11];
  const float* bo = (const float*)d_in[12];

  // Workspace carve-up (fully rewritten every call; 40 MB total):
  const size_t NELT = (size_t)BB * HH * NQS * DKh;  // 4,194,304
  _Float16* qh = (_Float16*)d_ws;
  _Float16* kh = qh + NELT;
  _Float16* vh = kh + NELT;
  float* ctx = (float*)(vh + NELT);  // [B][NQ][H*DV] f32

  dim3 gp(DM / 128, (BB * NQS) / 128);  // (8, 32)
  gemm_kernel<0><<<gp, 256, 0, stream>>>(queries, Wq, bq, qh, nullptr);
  gemm_kernel<0><<<gp, 256, 0, stream>>>(keys,    Wk, bk, kh, nullptr);
  gemm_kernel<0><<<gp, 256, 0, stream>>>(values,  Wv, bv, vh, nullptr);

  dim3 ga(NQS / 128, HH, BB);  // (16, 8, 2)
  attn_kernel<<<ga, 256, 0, stream>>>(qh, kh, vh, aweight, amask, ctx);

  gemm_kernel<1><<<gp, 256, 0, stream>>>(ctx, Wo, bo, nullptr, (float*)d_out);
}
